// GELU60_17566416240707
// MI455X (gfx1250) — matrix operand order repaired and checked by reading, not compiled, not run on hardware
//
#include <hip/hip_runtime.h>
#include <math.h>

#define B_  8
#define T_  2048
#define D_  1024
#define BM  64          // rows per workgroup (4 waves x 16-row tiles)
#define BN  32          // cols per LDS chunk (2 WMMA tiles per wave)
#define LDA 1032        // padded f16 row stride (+8 -> 4-bank rotation, conflict-free)

typedef __attribute__((ext_vector_type(16))) _Float16 v16h;
typedef __attribute__((ext_vector_type(8)))  _Float16 v8h;
typedef __attribute__((ext_vector_type(8)))  float    v8f;

union Frag16 { v16h v; v8h h8[2]; };

// 16B global -> LDS copy via the CDNA5 async path (ASYNCcnt-tracked, no VGPR
// round trip). Inline asm: the ROCm builtin's signature differs from upstream,
// but the ISA mnemonic/operand forms are fixed.
//   global_load_async_to_lds_b128 vdst(LDS byte offset), vaddr(64b), off
__device__ __forceinline__ void copy_b128(const _Float16* __restrict__ g,
                                          _Float16* __restrict__ l) {
    // flat LDS pointer: low 32 bits are the workgroup-relative LDS offset
    unsigned lds_off = (unsigned)(unsigned long long)(void*)l;
    unsigned long long ga = (unsigned long long)(const void*)g;
    asm volatile("global_load_async_to_lds_b128 %0, %1, off"
                 :
                 : "v"(lds_off), "v"(ga)
                 : "memory");
}

__device__ __forceinline__ void wait_async() {
    asm volatile("s_wait_asynccnt 0x0" ::: "memory");
}

__device__ __forceinline__ float gelu_f(float x) {
    const float c = 0.7978845608028654f;   // sqrt(2/pi)
    float inner = c * (x + 0.044715f * x * x * x);
    return 0.5f * x * (1.0f + tanhf(inner));
}

// ---------------------------------------------------------------------------
// Kernel 0: 1 / max(||ema||, 1e-12)
// ---------------------------------------------------------------------------
__global__ __launch_bounds__(256) void k_ema_rinv(const float* __restrict__ ema,
                                                  float* __restrict__ rinv_ema) {
    __shared__ float sred[8];
    int tid = threadIdx.x;
    int d = tid * 4;
    float4 e = *(const float4*)(ema + d);
    float ss = e.x * e.x + e.y * e.y + e.z * e.z + e.w * e.w;
    #pragma unroll
    for (int off = 16; off >= 1; off >>= 1) ss += __shfl_down(ss, off, 32);
    int wave = tid >> 5, lane = tid & 31;
    if (lane == 0) sred[wave] = ss;
    __syncthreads();
    if (tid == 0) {
        float s = 0.f;
        #pragma unroll
        for (int w = 0; w < 8; ++w) s += sred[w];
        rinv_ema[0] = 1.0f / fmaxf(sqrtf(s), 1e-12f);
    }
}

// ---------------------------------------------------------------------------
// Kernel 1: per row -> gelu, L2-normalize to f16 out_norm, and gelu.ema dot
// ---------------------------------------------------------------------------
__global__ __launch_bounds__(256) void k_gelu_norm(const float* __restrict__ x,
                                                   const float* __restrict__ ema,
                                                   _Float16* __restrict__ onorm,
                                                   float* __restrict__ emadot) {
    __shared__ float sred[16];
    int row = blockIdx.x;
    int tid = threadIdx.x;
    size_t rb = (size_t)row * D_;
    int d = tid * 4;

    float4 xv = *(const float4*)(x + rb + d);
    float g0 = gelu_f(xv.x), g1 = gelu_f(xv.y), g2 = gelu_f(xv.z), g3 = gelu_f(xv.w);
    float4 ev = *(const float4*)(ema + d);

    float ss = g0 * g0 + g1 * g1 + g2 * g2 + g3 * g3;
    float dp = g0 * ev.x + g1 * ev.y + g2 * ev.z + g3 * ev.w;
    #pragma unroll
    for (int off = 16; off >= 1; off >>= 1) {
        ss += __shfl_down(ss, off, 32);
        dp += __shfl_down(dp, off, 32);
    }
    int wave = tid >> 5, lane = tid & 31;
    if (lane == 0) { sred[wave] = ss; sred[8 + wave] = dp; }
    __syncthreads();
    if (tid == 0) {
        float s = 0.f, p = 0.f;
        #pragma unroll
        for (int w = 0; w < 8; ++w) { s += sred[w]; p += sred[8 + w]; }
        sred[0] = 1.0f / fmaxf(sqrtf(s), 1e-12f);
        sred[8] = p;
    }
    __syncthreads();
    float rinv = sred[0];

    union { _Float16 h[4]; uint2 u; } pk;
    pk.h[0] = (_Float16)(g0 * rinv);
    pk.h[1] = (_Float16)(g1 * rinv);
    pk.h[2] = (_Float16)(g2 * rinv);
    pk.h[3] = (_Float16)(g3 * rinv);
    *(uint2*)(onorm + rb + d) = pk.u;

    if (tid == 0) emadot[row] = sred[8] * rinv;   // * rinv_ema applied later
}

// ---------------------------------------------------------------------------
// Kernel 2: streaming causal row-max of out_norm @ out_norm^T via WMMA f16.
// grid (T/BM, B), 128 threads = 4 waves; wave w owns rows [r0+16w, r0+16w+16).
// Each wave computes BOTH 16-col tiles of the BN=32 chunk with a shared A
// fragment (2 accumulators). B chunks are double-buffered and staged with
// global_load_async_to_lds_b128, overlapped with the K-loop.
// ---------------------------------------------------------------------------
__global__ __launch_bounds__(128) void k_simmax(const _Float16* __restrict__ onorm,
                                                float* __restrict__ seqmax) {
    __shared__ _Float16 Alds[BM * LDA];        // 64 x 1024 (padded)  ~129 KB
    __shared__ _Float16 Blds[2][BN * LDA];     // 2 x 32 x 1024       ~129 KB

    const int tid  = threadIdx.x;
    const int lane = tid & 31;
    const int wave = tid >> 5;                 // == rowtile 0..3
    const int b  = blockIdx.y;
    const int r0 = blockIdx.x * BM;
    const size_t base = (size_t)b * T_ * D_;

    const int n     = lane & 15;               // tile col / A row within tile
    const int khalf = lane >> 4;               // which K-half this lane carries
    const int moff  = khalf * 8;               // C/D: lanes>=16 hold rows 8..15

    // stage A panel (BM rows) and B chunk 0 (uniform trip counts -> EXEC full)
    for (int i = tid; i < BM * (D_ / 8); i += 128) {
        int r = i >> 7;                        // D_/8 == 128
        int c = (i & 127) * 8;
        copy_b128(&onorm[base + (size_t)(r0 + r) * D_ + c], &Alds[r * LDA + c]);
    }
    for (int i = tid; i < BN * (D_ / 8); i += 128) {
        int r = i >> 7;
        int c = (i & 127) * 8;
        copy_b128(&onorm[base + (size_t)r * D_ + c], &Blds[0][r * LDA + c]);
    }
    wait_async();
    __syncthreads();

    const int nc   = (r0 >> 5) + 2;            // chunks cover cols [0, r0+BM)
    const int arow = wave * 16 + n;

    float pm[8];
    #pragma unroll
    for (int i = 0; i < 8; ++i) pm[i] = -2.0f;

    for (int cc = 0; cc < nc; ++cc) {
        const int buf = cc & 1;
        const int c0  = cc * BN;

        // issue async staging of the NEXT chunk into the other buffer
        // (that buffer was last read in iter cc-1; the trailing barrier of
        //  iter cc-1 guarantees everyone is done with it)
        if (cc + 1 < nc) {
            const int c1 = c0 + BN;
            for (int i = tid; i < BN * (D_ / 8); i += 128) {
                int r = i >> 7;
                int c = (i & 127) * 8;
                copy_b128(&onorm[base + (size_t)(c1 + r) * D_ + c],
                          &Blds[buf ^ 1][r * LDA + c]);
            }
        }

        const _Float16* Bb = Blds[buf];
        v8f acc0 = {0.f, 0.f, 0.f, 0.f, 0.f, 0.f, 0.f, 0.f};
        v8f acc1 = {0.f, 0.f, 0.f, 0.f, 0.f, 0.f, 0.f, 0.f};

        #pragma unroll 4
        for (int kb = 0; kb < D_; kb += 32) {
            Frag16 a, b0, b1;
            const int klo = kb + khalf * 8;        // A 16x32 f16 layout
            a.h8[0] = *(const v8h*)&Alds[arow * LDA + klo];
            a.h8[1] = *(const v8h*)&Alds[arow * LDA + klo + 16];
            const int kbb = kb + khalf * 8;        // B 32x16 f16 layout
            b0.h8[0] = *(const v8h*)&Bb[n * LDA + kbb];
            b0.h8[1] = *(const v8h*)&Bb[n * LDA + kbb + 16];
            b1.h8[0] = *(const v8h*)&Bb[(16 + n) * LDA + kbb];
            b1.h8[1] = *(const v8h*)&Bb[(16 + n) * LDA + kbb + 16];
            acc0 = __builtin_amdgcn_wmma_f32_16x16x32_f16(
                false, a.v, false, b0.v, (short)0, acc0, false, false);
            acc1 = __builtin_amdgcn_wmma_f32_16x16x32_f16(
                false, a.v, false, b1.v, (short)0, acc1, false, false);
        }

        // strict-causal mask + running row max (cols c0+n and c0+16+n)
        const int gc0 = c0 + n;
        const int gc1 = c0 + 16 + n;
        #pragma unroll
        for (int i = 0; i < 8; ++i) {
            const int gr = r0 + wave * 16 + moff + i;
            float v0 = (gc0 < gr) ? acc0[i] : -2.0f;
            float v1 = (gc1 < gr) ? acc1[i] : -2.0f;
            pm[i] = fmaxf(pm[i], fmaxf(v0, v1));
        }

        wait_async();          // next chunk's panel has landed in LDS
        __syncthreads();       // all waves done reading current buffer
    }

    // reduce over the 16 lanes of each half (xor masks stay within the half)
    #pragma unroll
    for (int i = 0; i < 8; ++i) {
        #pragma unroll
        for (int off = 8; off >= 1; off >>= 1)
            pm[i] = fmaxf(pm[i], __shfl_xor(pm[i], off, 32));
    }
    // wave exclusively owns its 16 rows: lanes 0 / 16 write 8 rows each
    if (n == 0) {
        #pragma unroll
        for (int i = 0; i < 8; ++i)
            seqmax[(size_t)b * T_ + r0 + wave * 16 + moff + i] = pm[i];
    }
}

// ---------------------------------------------------------------------------
// Kernel 3: gate and write output (recompute gelu -> x read twice, out once)
// ---------------------------------------------------------------------------
__global__ __launch_bounds__(256) void k_gate(const float* __restrict__ x,
                                              const float* __restrict__ emadot,
                                              const float* __restrict__ seqmax,
                                              const float* __restrict__ rinv_ema,
                                              const float* __restrict__ log_tau,
                                              const float* __restrict__ log_blend,
                                              const float* __restrict__ logit_blend_seq,
                                              float* __restrict__ out) {
    int row = blockIdx.x;
    int tid = threadIdx.x;
    int t = row & (T_ - 1);

    float tau   = expf(log_tau[0]);
    float alpha = 1.0f / (1.0f + expf(-log_blend[0]));
    float wseq  = 1.0f / (1.0f + expf(-logit_blend_seq[0]));

    float es  = emadot[row] * rinv_ema[0];
    float sq  = (t == 0) ? es : seqmax[row];
    float fam = wseq * sq + (1.0f - wseq) * es;
    float gate = 1.0f - alpha + alpha * expf(-tau * fam);

    size_t rb = (size_t)row * D_;
    int d = tid * 4;
    float4 xv = *(const float4*)(x + rb + d);
    float4 o;
    o.x = gelu_f(xv.x) * gate;
    o.y = gelu_f(xv.y) * gate;
    o.z = gelu_f(xv.z) * gate;
    o.w = gelu_f(xv.w) * gate;
    *(float4*)(out + rb + d) = o;
}

// ---------------------------------------------------------------------------
extern "C" void kernel_launch(void* const* d_in, const int* in_sizes, int n_in,
                              void* d_out, int out_size, void* d_ws, size_t ws_size,
                              hipStream_t stream) {
    const float* x               = (const float*)d_in[0];
    const float* ema             = (const float*)d_in[1];
    // d_in[2] = logit_decay (unused by the reference)
    const float* log_tau         = (const float*)d_in[3];
    const float* log_blend       = (const float*)d_in[4];
    const float* logit_blend_seq = (const float*)d_in[5];
    float* out = (float*)d_out;

    // workspace layout
    _Float16* onorm = (_Float16*)d_ws;                              // B*T*D f16 (32 MB)
    char* p = (char*)d_ws + (size_t)B_ * T_ * D_ * sizeof(_Float16);
    float* emadot   = (float*)p;                                    // B*T
    float* seqmax   = emadot + (size_t)B_ * T_;                     // B*T
    float* rinv_ema = seqmax + (size_t)B_ * T_;                     // 1

    k_ema_rinv<<<1, 256, 0, stream>>>(ema, rinv_ema);
    k_gelu_norm<<<B_ * T_, 256, 0, stream>>>(x, ema, onorm, emadot);
    k_simmax<<<dim3(T_ / BM, B_), 128, 0, stream>>>(onorm, seqmax);
    k_gate<<<B_ * T_, 256, 0, stream>>>(x, emadot, seqmax, rinv_ema,
                                        log_tau, log_blend, logit_blend_seq, out);
}
